// MultiAttention_13726715478235
// MI455X (gfx1250) — matrix operand order repaired and checked by reading, not compiled, not run on hardware
//
#include <hip/hip_runtime.h>
#include <hip/hip_bf16.h>
#include <stdint.h>

// ---------------------------------------------------------------------------
// MultiHeadAttention (B=1, S=2048, D=1024, H=16, HD=64) for gfx1250 (MI455X)
// - All matmuls via V_WMMA_F32_16X16X32_F16 (wave32 fragments per CDNA5 ISA)
// - Shared tiles double-buffered in LDS via GLOBAL_LOAD_ASYNC_TO_LDS_B128,
//   drained with partial S_WAIT_ASYNCCNT so the next tile's copies fly
//   under the current tile's WMMAs.
// ---------------------------------------------------------------------------

typedef __attribute__((ext_vector_type(16))) _Float16 v16h;
typedef __attribute__((ext_vector_type(8)))  _Float16 v8h;
typedef __attribute__((ext_vector_type(4)))  _Float16 v4h;
typedef __attribute__((ext_vector_type(8)))  float    v8f;

#define SEQ 2048
#define DMODEL 1024
#define NHEAD 16
#define HDIM 64

static __device__ __forceinline__ v8f wmma_f16(v16h a, v16h b, v8f c) {
  return __builtin_amdgcn_wmma_f32_16x16x32_f16(false, a, false, b, (short)0, c,
                                                false, false);
}

// A-fragment (16x32 f16, MxK): lane (m = lane&15, hi = lane>>4) holds
// K = k0+hi*8..+7 (VGPR0..3) and K = k0+16+hi*8..+7 (VGPR4..7).
static __device__ __forceinline__ v16h load_a_frag(const _Float16* p) {
  v8h lo = *(const v8h*)(p);
  v8h hh = *(const v8h*)(p + 16);
  v16h r;
#pragma unroll
  for (int i = 0; i < 8; ++i) { r[i] = lo[i]; r[i + 8] = hh[i]; }
  return r;
}

// Async 16-byte global -> LDS copy (CDNA5, tracked by ASYNCcnt).
// VDST holds the LDS byte address = low 32 bits of the generic pointer.
static __device__ __forceinline__ void async_ld16(_Float16* lds_dst,
                                                  const _Float16* gsrc) {
  unsigned int l = (unsigned int)(uintptr_t)lds_dst;
  asm volatile("global_load_async_to_lds_b128 %0, %1, off"
               :: "v"(l), "v"(gsrc)
               : "memory");
}
// Partial drains: async ops complete in order, so waiting ASYNCcnt<=N leaves
// the most recent N copies in flight.
#define WAIT_ASYNC_LE1() asm volatile("s_wait_asynccnt 0x1" ::: "memory")
#define WAIT_ASYNC_LE2() asm volatile("s_wait_asynccnt 0x2" ::: "memory")

// ---------------------------------------------------------------------------
// f32 -> f16 conversion (vectorized x4)
// ---------------------------------------------------------------------------
__global__ void cvt_f32_f16(const float4* __restrict__ src,
                            _Float16* __restrict__ dst, int n4) {
  int i = blockIdx.x * blockDim.x + threadIdx.x;
  if (i < n4) {
    float4 v = src[i];
    v4h h;
    h[0] = (_Float16)v.x; h[1] = (_Float16)v.y;
    h[2] = (_Float16)v.z; h[3] = (_Float16)v.w;
    *(v4h*)(dst + 4 * (size_t)i) = h;
  }
}

// ---------------------------------------------------------------------------
// QKV projection + bias + fused RoPE.  qkv = x @ qkv_w^T + b.
// Block: 8 waves, one 64-col tile (== one head slice), 256 rows.
// Wave: 32 rows x 64 cols (2 A-frags x 4 B-frags = 8 WMMAs / k-step).
// Weight tile (64 rows x 32 k = 4KB) double-buffered in LDS (async copies).
// Q,K: RoPE'd, stored per-head row-major [h][s][64].
// V:   stored per-head transposed [h][d][s] so P@V B-fragments are contiguous.
// ---------------------------------------------------------------------------
__global__ __launch_bounds__(256) void qkv_rope_kernel(
    const _Float16* __restrict__ xh, const _Float16* __restrict__ wh,
    const float* __restrict__ bias, _Float16* __restrict__ Qh,
    _Float16* __restrict__ Kh, _Float16* __restrict__ Vt) {
  __shared__ __align__(16) _Float16 Bl[2][64 * 32];  // 2 x 4KB weight tiles
  const int t    = threadIdx.x;
  const int wib  = t >> 5;
  const int lane = t & 31;
  const int NT   = (3 * DMODEL) / 64;                // 48 col tiles
  const int col0 = (blockIdx.x % NT) * 64;
  const int row0 = (blockIdx.x / NT) * 256 + wib * 32;
  const int m = lane & 15, hi = lane >> 4;

  const _Float16* arow0 = xh + (size_t)(row0 + m) * DMODEL + hi * 8;
  const _Float16* arow1 = arow0 + (size_t)16 * DMODEL;
  const _Float16* wrow  = wh + (size_t)(col0 + (t >> 2)) * DMODEL + (t & 3) * 8;
  _Float16* blr[2] = { Bl[0] + (t >> 2) * 32 + (t & 3) * 8,
                       Bl[1] + (t >> 2) * 32 + (t & 3) * 8 };

  v8f c[2][4] = {};
  async_ld16(blr[0], wrow);                // stage k-step 0 into buffer 0
  v16h a0 = load_a_frag(arow0);
  v16h a1 = load_a_frag(arow1);

  const int NK = DMODEL / 32;              // 32 k-steps
  for (int i = 0; i < NK; ++i) {
    const int k0 = i * 32;
    const int kn = (i + 1 < NK) ? k0 + 32 : k0;  // clamped (dummy last stage)
    __syncthreads();                       // other buffer's reads are done
    async_ld16(blr[(i + 1) & 1], wrow + kn);     // stage next tile
    WAIT_ASYNC_LE1();                      // drain current tile only
    __syncthreads();

    v16h a0c = a0, a1c = a1;
    if (i + 1 < NK) {                      // prefetch next A under the WMMAs
      a0 = load_a_frag(arow0 + k0 + 32);
      a1 = load_a_frag(arow1 + k0 + 32);
    }
    const _Float16* Bc = Bl[i & 1];
#pragma unroll
    for (int j = 0; j < 4; ++j) {
      v16h b = load_a_frag(Bc + (j * 16 + m) * 32 + hi * 8);
      c[0][j] = wmma_f16(a0c, b, c[0][j]);
      c[1][j] = wmma_f16(a1c, b, c[1][j]);
    }
  }

  // bias (per output column)
#pragma unroll
  for (int j = 0; j < 4; ++j) {
    float bj = bias[col0 + j * 16 + m];
#pragma unroll
    for (int rr = 0; rr < 2; ++rr)
#pragma unroll
      for (int r = 0; r < 8; ++r) c[rr][j][r] += bj;
  }

  const int section = col0 >> 10;          // 0 = Q, 1 = K, 2 = V
  const int h = (col0 & (DMODEL - 1)) >> 6;

  if (section < 2) {
    _Float16* dst = (section == 0 ? Qh : Kh) + (size_t)h * SEQ * HDIM;
#pragma unroll
    for (int rr = 0; rr < 2; ++rr) {
      int rbase = row0 + rr * 16;
#pragma unroll
      for (int jj = 0; jj < 2; ++jj) {     // RoPE pair (d, d+32) = frag (jj, jj+2)
        int fi = jj * 16 + m;
        float invf = __powf(10000.f, -(float)fi * (1.0f / 32.0f));
#pragma unroll
        for (int r = 0; r < 8; ++r) {
          int s = rbase + hi * 8 + r;
          float sn, cs;
          __sincosf((float)s * invf, &sn, &cs);
          float t1 = c[rr][jj][r], t2 = c[rr][jj + 2][r];
          dst[(size_t)s * HDIM + fi]      = (_Float16)(t1 * cs - t2 * sn);
          dst[(size_t)s * HDIM + 32 + fi] = (_Float16)(t1 * sn + t2 * cs);
        }
      }
    }
  } else {
    _Float16* dst = Vt + (size_t)h * HDIM * SEQ;
#pragma unroll
    for (int rr = 0; rr < 2; ++rr)
#pragma unroll
      for (int j = 0; j < 4; ++j) {
        int d = j * 16 + m;
#pragma unroll
        for (int r = 0; r < 8; ++r) {
          int s = row0 + rr * 16 + hi * 8 + r;
          dst[(size_t)d * SEQ + s] = (_Float16)c[rr][j][r];
        }
      }
  }
}

// ---------------------------------------------------------------------------
// Flash attention. Block = 8 waves on 8 consecutive query tiles of ONE head;
// 32-key K/V tiles (4KB each) double-buffered in LDS via async copies and
// shared by all 8 waves.  Per key tile: 4 WMMAs for S = Q K^T, online softmax
// (16-lane shfl reductions), P transposed C->A layout through per-wave LDS,
// 4 WMMAs for P V.
// ---------------------------------------------------------------------------
__global__ __launch_bounds__(256) void flash_kernel(
    const _Float16* __restrict__ Qh, const _Float16* __restrict__ Kh,
    const _Float16* __restrict__ Vt, _Float16* __restrict__ ctxh) {
  __shared__ __align__(16) _Float16 Klds[2][32 * 64];    // 2 x 4KB
  __shared__ __align__(16) _Float16 Vlds[2][64 * 32];    // 2 x 4KB
  __shared__ __align__(16) _Float16 Plds[8][16 * 40];    // 10KB per-wave P
  const int t    = threadIdx.x;
  const int wib  = t >> 5;
  const int lane = t & 31;
  const int h  = blockIdx.x >> 4;          // 16 blocks per head
  const int qt = (blockIdx.x & 15) * 8 + wib;
  const int m = lane & 15, hi = lane >> 4;
  const int row0 = qt * 16;

  const _Float16* Qb = Qh + (size_t)h * SEQ * HDIM;
  const _Float16* Kb = Kh + (size_t)h * SEQ * HDIM;
  const _Float16* Vb = Vt + (size_t)h * HDIM * SEQ;
  _Float16* P = Plds[wib];

  // staging addresses: K tile row = t/8 (128B rows), V tile row = t/4 (64B rows)
  const _Float16* kg = Kb + (size_t)(t >> 3) * HDIM + (t & 7) * 8;
  const _Float16* vg = Vb + (size_t)(t >> 2) * SEQ + (t & 3) * 8;
  _Float16* kl[2] = { Klds[0] + (t >> 3) * 64 + (t & 7) * 8,
                      Klds[1] + (t >> 3) * 64 + (t & 7) * 8 };
  _Float16* vl[2] = { Vlds[0] + (t >> 2) * 32 + (t & 3) * 8,
                      Vlds[1] + (t >> 2) * 32 + (t & 3) * 8 };

  // stage key-tile 0 into buffer 0
  async_ld16(kl[0], kg);
  async_ld16(vl[0], vg);

  const v16h aq0 = load_a_frag(Qb + (size_t)(row0 + m) * HDIM + hi * 8);
  const v16h aq1 = load_a_frag(Qb + (size_t)(row0 + m) * HDIM + 32 + hi * 8);

  float mi[8], li[8];
  v8f o[4] = {};
#pragma unroll
  for (int r = 0; r < 8; ++r) { mi[r] = -1e30f; li[r] = 0.0f; }

  const int NKT = SEQ / 32;                // 64 key tiles
  for (int kt = 0; kt < NKT; ++kt) {
    const int kbase = kt * 32;
    const int knext = (kt + 1 < NKT) ? kbase + 32 : kbase;  // clamped
    __syncthreads();                       // other buffer's reads are done
    async_ld16(kl[(kt + 1) & 1], kg + (size_t)knext * HDIM);
    async_ld16(vl[(kt + 1) & 1], vg + knext);
    WAIT_ASYNC_LE2();                      // drain current tile's 2 copies
    __syncthreads();

    const _Float16* Kc = Klds[kt & 1];
    const _Float16* Vc = Vlds[kt & 1];

    v8f s0 = {}, s1 = {};
    {
      v16h b = load_a_frag(Kc + m * 64 + hi * 8);
      s0 = wmma_f16(aq0, b, s0);
      b = load_a_frag(Kc + m * 64 + 32 + hi * 8);
      s0 = wmma_f16(aq1, b, s0);
      b = load_a_frag(Kc + (16 + m) * 64 + hi * 8);
      s1 = wmma_f16(aq0, b, s1);
      b = load_a_frag(Kc + (16 + m) * 64 + 32 + hi * 8);
      s1 = wmma_f16(aq1, b, s1);
    }

    float pscale[8];
#pragma unroll
    for (int r = 0; r < 8; ++r) {
      float v0 = s0[r] * 0.125f, v1 = s1[r] * 0.125f;   // 1/sqrt(64)
      float mx = fmaxf(v0, v1);
#pragma unroll
      for (int d = 8; d >= 1; d >>= 1) mx = fmaxf(mx, __shfl_xor(mx, d, 32));
      float mn = fmaxf(mi[r], mx);
      float p0 = __expf(v0 - mn), p1 = __expf(v1 - mn);
      float sum = p0 + p1;
#pragma unroll
      for (int d = 8; d >= 1; d >>= 1) sum += __shfl_xor(sum, d, 32);
      float sc = __expf(mi[r] - mn);
      li[r] = li[r] * sc + sum;
      mi[r] = mn;
      pscale[r] = sc;
      const int rowp = hi * 8 + r;
      P[rowp * 40 + m]      = (_Float16)p0;
      P[rowp * 40 + 16 + m] = (_Float16)p1;
    }
#pragma unroll
    for (int j = 0; j < 4; ++j)
#pragma unroll
      for (int r = 0; r < 8; ++r) o[j][r] *= pscale[r];

    __syncthreads();                       // P stores drained (dscnt)
    v16h ap = load_a_frag(P + m * 40 + hi * 8);   // P as A-fragment (K=32 keys)
#pragma unroll
    for (int j = 0; j < 4; ++j) {
      v16h b = load_a_frag(Vc + (j * 16 + m) * 32 + hi * 8);
      o[j] = wmma_f16(ap, b, o[j]);
    }
  }

#pragma unroll
  for (int j = 0; j < 4; ++j)
#pragma unroll
    for (int r = 0; r < 8; ++r) {
      int s = row0 + hi * 8 + r;
      float val = o[j][r] / li[r];
      ctxh[(size_t)s * DMODEL + h * HDIM + j * 16 + m] = (_Float16)val;
    }
}

// ---------------------------------------------------------------------------
// Output projection + bias + residual:  out = ctx @ out_w^T + b + x  (fp32)
// Same double-buffered structure as QKV.
// ---------------------------------------------------------------------------
__global__ __launch_bounds__(256) void outproj_kernel(
    const _Float16* __restrict__ ctxh, const _Float16* __restrict__ owh,
    const float* __restrict__ ob, const float* __restrict__ x,
    float* __restrict__ out) {
  __shared__ __align__(16) _Float16 Bl[2][64 * 32];
  const int t    = threadIdx.x;
  const int wib  = t >> 5;
  const int lane = t & 31;
  const int NT   = DMODEL / 64;            // 16 col tiles
  const int col0 = (blockIdx.x % NT) * 64;
  const int row0 = (blockIdx.x / NT) * 256 + wib * 32;
  const int m = lane & 15, hi = lane >> 4;

  const _Float16* arow0 = ctxh + (size_t)(row0 + m) * DMODEL + hi * 8;
  const _Float16* arow1 = arow0 + (size_t)16 * DMODEL;
  const _Float16* wrow  = owh + (size_t)(col0 + (t >> 2)) * DMODEL + (t & 3) * 8;
  _Float16* blr[2] = { Bl[0] + (t >> 2) * 32 + (t & 3) * 8,
                       Bl[1] + (t >> 2) * 32 + (t & 3) * 8 };

  v8f c[2][4] = {};
  async_ld16(blr[0], wrow);
  v16h a0 = load_a_frag(arow0);
  v16h a1 = load_a_frag(arow1);

  const int NK = DMODEL / 32;
  for (int i = 0; i < NK; ++i) {
    const int k0 = i * 32;
    const int kn = (i + 1 < NK) ? k0 + 32 : k0;
    __syncthreads();
    async_ld16(blr[(i + 1) & 1], wrow + kn);
    WAIT_ASYNC_LE1();
    __syncthreads();

    v16h a0c = a0, a1c = a1;
    if (i + 1 < NK) {
      a0 = load_a_frag(arow0 + k0 + 32);
      a1 = load_a_frag(arow1 + k0 + 32);
    }
    const _Float16* Bc = Bl[i & 1];
#pragma unroll
    for (int j = 0; j < 4; ++j) {
      v16h b = load_a_frag(Bc + (j * 16 + m) * 32 + hi * 8);
      c[0][j] = wmma_f16(a0c, b, c[0][j]);
      c[1][j] = wmma_f16(a1c, b, c[1][j]);
    }
  }

#pragma unroll
  for (int j = 0; j < 4; ++j) {
    int n = col0 + j * 16 + m;
    float bj = ob[n];
#pragma unroll
    for (int rr = 0; rr < 2; ++rr)
#pragma unroll
      for (int r = 0; r < 8; ++r) {
        int s = row0 + rr * 16 + hi * 8 + r;
        out[(size_t)s * DMODEL + n] = c[rr][j][r] + bj + x[(size_t)s * DMODEL + n];
      }
  }
}

// ---------------------------------------------------------------------------
extern "C" void kernel_launch(void* const* d_in, const int* in_sizes, int n_in,
                              void* d_out, int out_size, void* d_ws,
                              size_t ws_size, hipStream_t stream) {
  const float* x     = (const float*)d_in[0];
  const float* qkv_w = (const float*)d_in[1];
  const float* qkv_b = (const float*)d_in[2];
  const float* out_w = (const float*)d_in[3];
  const float* out_b = (const float*)d_in[4];
  float* out = (float*)d_out;

  char* ws = (char*)d_ws;
  _Float16* xh   = (_Float16*)ws; ws += (size_t)SEQ * DMODEL * 2;
  _Float16* wh   = (_Float16*)ws; ws += (size_t)3 * DMODEL * DMODEL * 2;
  _Float16* owh  = (_Float16*)ws; ws += (size_t)DMODEL * DMODEL * 2;
  _Float16* Qh   = (_Float16*)ws; ws += (size_t)SEQ * DMODEL * 2;
  _Float16* Kh   = (_Float16*)ws; ws += (size_t)SEQ * DMODEL * 2;
  _Float16* Vt   = (_Float16*)ws; ws += (size_t)SEQ * DMODEL * 2;
  _Float16* ctxh = (_Float16*)ws; ws += (size_t)SEQ * DMODEL * 2;
  // total workspace: ~28 MB

  {
    int n4 = SEQ * DMODEL / 4;
    cvt_f32_f16<<<(n4 + 255) / 256, 256, 0, stream>>>((const float4*)x, xh, n4);
    n4 = 3 * DMODEL * DMODEL / 4;
    cvt_f32_f16<<<(n4 + 255) / 256, 256, 0, stream>>>((const float4*)qkv_w, wh, n4);
    n4 = DMODEL * DMODEL / 4;
    cvt_f32_f16<<<(n4 + 255) / 256, 256, 0, stream>>>((const float4*)out_w, owh, n4);
  }

  // QKV + RoPE: (2048/256 row blocks) * 48 col tiles = 384 blocks of 8 waves
  qkv_rope_kernel<<<384, 256, 0, stream>>>(xh, wh, qkv_b, Qh, Kh, Vt);

  // Flash attention: 16 heads * 16 blocks (8 q-tiles each) = 256 blocks
  flash_kernel<<<256, 256, 0, stream>>>(Qh, Kh, Vt, ctxh);

  // Output projection: 8 row blocks * 16 col tiles = 128 blocks
  outproj_kernel<<<128, 256, 0, stream>>>(ctxh, owh, out_b, x, out);
}
